// GPTLanguageModel_34316788695335
// MI455X (gfx1250) — matrix operand order repaired and checked by reading, not compiled, not run on hardware
//
#include <hip/hip_runtime.h>
#include <hip/hip_bf16.h>

// ---------------------------------------------------------------------------
// GPT forward for MI455X (gfx1250), wave32 + WMMA f16 (f32 accumulate).
// ---------------------------------------------------------------------------

typedef __attribute__((ext_vector_type(16))) _Float16 v16h;
typedef __attribute__((ext_vector_type(4)))  _Float16 v4h;
typedef __attribute__((ext_vector_type(8)))  float    v8f;

#define TB 4
#define TT 1024
#define TE 1024
#define THN 16
#define THD 64
#define TL 6
#define TFF 4096
#define TV 32000
#define TM (TB * TT)   // 4096 rows of the activation matrix

__device__ __forceinline__ v8f zero8() {
  v8f z;
#pragma unroll
  for (int i = 0; i < 8; ++i) z[i] = 0.f;
  return z;
}

__device__ __forceinline__ v4h cvt4(float4 f) {
  v4h h;
  h[0] = (_Float16)f.x; h[1] = (_Float16)f.y;
  h[2] = (_Float16)f.z; h[3] = (_Float16)f.w;
  return h;
}

// A-fragment (16x32 f16, MxK). Lanes 0-15: M=0..15, K 0..7 & 16..23;
// lanes 16-31: K 8..15 & 24..31; 2 consecutive K per VGPR.
__device__ __forceinline__ v16h load_a_frag(const _Float16* __restrict__ lds,
                                            int ld, int lane) {
  const int m  = lane & 15;
  const int kh = (lane >> 4) * 8;
  const _Float16* row = lds + m * ld;
  v16h a;
#pragma unroll
  for (int j = 0; j < 4; ++j) {
    a[2 * j]     = row[kh + 2 * j];
    a[2 * j + 1] = row[kh + 2 * j + 1];
    a[8 + 2 * j]     = row[16 + kh + 2 * j];
    a[8 + 2 * j + 1] = row[16 + kh + 2 * j + 1];
  }
  return a;
}

// B-fragment (32x16 f16, KxN), LDS holds B transposed: ldsT[n][k] (k contiguous).
__device__ __forceinline__ v16h load_b_frag(const _Float16* __restrict__ ldsT,
                                            int ld, int lane) {
  const int n    = lane & 15;
  const int koff = (lane >> 4) * 8;
  const _Float16* col = ldsT + n * ld;
  v16h b;
#pragma unroll
  for (int j = 0; j < 8; ++j) {
    b[j]     = col[koff + j];
    b[8 + j] = col[16 + koff + j];
  }
  return b;
}

__device__ __forceinline__ v8f wmma_f16(v16h a, v16h b, v8f c) {
  return __builtin_amdgcn_wmma_f32_16x16x32_f16(false, a, false, b,
                                                (short)0, c, false, false);
}

// ---------------------------------------------------------------------------
// Generic GEMM: C[M,N] = A[M,K] @ B[K,N] (+bias +residual, optional ReLU).
// 256 threads = 8 waves; block tile 256x128, per-wave tile 64x64 (4x4 WMMA
// fragments -> 16 wmma per k-step vs 8 fragment loads). BK=32, double-buffered
// LDS. Staging uses per-thread pointers advanced by fixed strides (no 64-bit
// multiplies in the pipelined loop) with float4 global loads -> f16 LDS.
// flags: 1=bias, 2=relu, 4=residual.
// M % 256 == 0, N % 128 == 0, K % 32 == 0 (true for all GEMMs here).
// ---------------------------------------------------------------------------
__global__ void __launch_bounds__(256)
gemm_wmma(const float* __restrict__ A, const float* __restrict__ Bw,
          const float* __restrict__ bias, const float* __restrict__ residual,
          float* __restrict__ C, int M, int N, int K, int flags) {
  __shared__ _Float16 As[2][256][40];   // [buf][m][k]
  __shared__ _Float16 Bs[2][128][40];   // [buf][n][k]  (B transposed)

  const int tid  = threadIdx.x;
  const int lane = tid & 31;
  const int wave = tid >> 5;
  const int m0 = blockIdx.y * 256;
  const int n0 = blockIdx.x * 128;
  const int wm = (wave >> 1) * 64;   // 4 waves along M
  const int wn = (wave & 1) * 64;    // 2 waves along N

  v8f acc[4][4];
#pragma unroll
  for (int mi = 0; mi < 4; ++mi)
#pragma unroll
    for (int ni = 0; ni < 4; ++ni) acc[mi][ni] = zero8();

  // Per-thread staging pointers, advanced by constant strides each k-step.
  const int ar  = tid >> 3;           // A row within tile (0..31, step 32)
  const int ac4 = (tid & 7) * 4;      // A col group
  const int bk  = tid >> 5;           // B k within tile (0..7, step 8)
  const int bn4 = (tid & 31) * 4;     // B col group
  const float* ap = A + (size_t)(m0 + ar) * K + ac4;       // += 32 per k-step
  const float* bp = Bw + (size_t)bk * N + n0 + bn4;        // += 32*N per k-step
  const size_t aStride = (size_t)K * 32;                   // per unrolled j
  const size_t bStride = (size_t)N * 8;                    // per unrolled j

  auto stage = [&](int buf) {
    const float* a = ap;
#pragma unroll
    for (int j = 0; j < 8; ++j) {
      const float4 f = *reinterpret_cast<const float4*>(a);
      *reinterpret_cast<v4h*>(&As[buf][ar + j * 32][ac4]) = cvt4(f);
      a += aStride;
    }
    const float* b = bp;
#pragma unroll
    for (int j = 0; j < 4; ++j) {
      const float4 f = *reinterpret_cast<const float4*>(b);
      Bs[buf][bn4 + 0][bk + j * 8] = (_Float16)f.x;
      Bs[buf][bn4 + 1][bk + j * 8] = (_Float16)f.y;
      Bs[buf][bn4 + 2][bk + j * 8] = (_Float16)f.z;
      Bs[buf][bn4 + 3][bk + j * 8] = (_Float16)f.w;
      b += bStride;
    }
    ap += 32;                 // advance one k-step (32 floats along K)
    bp += (size_t)N * 32;     // advance one k-step (32 rows of B)
  };

  const int ksteps = K >> 5;
  stage(0);

  for (int kt = 0; kt < ksteps; ++kt) {
    __syncthreads();
    if (kt + 1 < ksteps) stage((kt + 1) & 1);
    if (kt + 2 < ksteps) {   // global_prefetch_b8 of the k+3 tile
      __builtin_prefetch(ap + 32, 0, 1);
      __builtin_prefetch(bp + (size_t)N * 32, 0, 1);
    }
    const int buf = kt & 1;

    v16h af[4], bf[4];
#pragma unroll
    for (int mi = 0; mi < 4; ++mi)
      af[mi] = load_a_frag(&As[buf][wm + mi * 16][0], 40, lane);
#pragma unroll
    for (int ni = 0; ni < 4; ++ni)
      bf[ni] = load_b_frag(&Bs[buf][wn + ni * 16][0], 40, lane);
#pragma unroll
    for (int mi = 0; mi < 4; ++mi)
#pragma unroll
      for (int ni = 0; ni < 4; ++ni)
        acc[mi][ni] = wmma_f16(af[mi], bf[ni], acc[mi][ni]);
  }

  // Epilogue. C layout: VGPR r, lanes 0-15 -> M=r, lanes 16-31 -> M=8+r.
  const int nl = lane & 15;
  const int mh = (lane >> 4) * 8;
#pragma unroll
  for (int mi = 0; mi < 4; ++mi)
#pragma unroll
    for (int ni = 0; ni < 4; ++ni) {
      const int col = n0 + wn + ni * 16 + nl;
      const float bcol = (flags & 1) ? bias[col] : 0.f;
      size_t base = (size_t)(m0 + wm + mi * 16 + mh) * N + col;
#pragma unroll
      for (int r = 0; r < 8; ++r) {
        float v = acc[mi][ni][r] + bcol;
        if (flags & 4) v += residual[base];
        if (flags & 2) v = fmaxf(v, 0.f);
        C[base] = v;
        base += N;
      }
    }
}

// ---------------------------------------------------------------------------
// Flash attention (causal): one workgroup = (b, h, 64 q-rows); 4 waves, each
// owning 16 q-rows. hd = 64. Online softmax with cross-lane reductions that
// match the 16x16 f32 C layout (each row lives in one 16-lane half).
// ---------------------------------------------------------------------------
__global__ void __launch_bounds__(128)
attn_kernel(const float* __restrict__ Q, const float* __restrict__ K,
            const float* __restrict__ Vv, float* __restrict__ O) {
  __shared__ _Float16 sQ[4][16][72];
  __shared__ _Float16 sP[4][16][72];
  __shared__ _Float16 sK[64][72];    // [key][dim]
  __shared__ _Float16 sVt[64][72];   // [dim][key]  (transposed)

  const int tid  = threadIdx.x;
  const int lane = tid & 31;
  const int wave = tid >> 5;
  const int b = blockIdx.x / THN;
  const int h = blockIdx.x % THN;
  const int qblk  = blockIdx.y;
  const int qrow0 = qblk * 64 + wave * 16;
  const size_t baseBT = (size_t)b * TT;
  const int col0 = h * THD;

  // stage this wave's Q tile (16 x 64) as float4 loads + packed f16 stores
  for (int i = lane; i < 16 * 16; i += 32) {
    const int r = i >> 4, d4 = (i & 15) * 4;
    const float4 f = *reinterpret_cast<const float4*>(
        &Q[(baseBT + qrow0 + r) * TE + col0 + d4]);
    *reinterpret_cast<v4h*>(&sQ[wave][r][d4]) = cvt4(f);
  }
  __syncthreads();
  const v16h aq0 = load_a_frag(&sQ[wave][0][0], 72, lane);
  const v16h aq1 = load_a_frag(&sQ[wave][0][32], 72, lane);

  v8f o[4];
#pragma unroll
  for (int nt = 0; nt < 4; ++nt) o[nt] = zero8();
  float mrun[8], lrun[8];
#pragma unroll
  for (int r = 0; r < 8; ++r) { mrun[r] = -1e30f; lrun[r] = 0.f; }

  const int nl = lane & 15;
  const int mh = (lane >> 4) * 8;

  for (int kb = 0; kb <= qblk; ++kb) {
    __syncthreads();
#pragma unroll
    for (int j = 0; j < 8; ++j) {
      const int i = tid + j * 128;
      const int kr = i >> 4, d4 = (i & 15) * 4;
      const size_t g = (baseBT + kb * 64 + kr) * TE + col0 + d4;
      const float4 kf = *reinterpret_cast<const float4*>(&K[g]);
      *reinterpret_cast<v4h*>(&sK[kr][d4]) = cvt4(kf);
      const float4 vf = *reinterpret_cast<const float4*>(&Vv[g]);
      sVt[d4 + 0][kr] = (_Float16)vf.x;
      sVt[d4 + 1][kr] = (_Float16)vf.y;
      sVt[d4 + 2][kr] = (_Float16)vf.z;
      sVt[d4 + 3][kr] = (_Float16)vf.w;
    }
    __syncthreads();

    // S = Q @ K^T  (16 q-rows x 64 keys), two K=32 steps over hd
    v8f s[4];
#pragma unroll
    for (int kt2 = 0; kt2 < 4; ++kt2) {
      v8f acc = zero8();
      v16h bk = load_b_frag(&sK[kt2 * 16][0], 72, lane);
      acc = wmma_f16(aq0, bk, acc);
      bk = load_b_frag(&sK[kt2 * 16][32], 72, lane);
      acc = wmma_f16(aq1, bk, acc);
      s[kt2] = acc;
    }

    // scale + causal mask
#pragma unroll
    for (int kt2 = 0; kt2 < 4; ++kt2)
#pragma unroll
      for (int r = 0; r < 8; ++r) {
        const int key = kb * 64 + kt2 * 16 + nl;
        const int qr  = qrow0 + mh + r;
        const float v = s[kt2][r] * 0.125f;  // 1/sqrt(64)
        s[kt2][r] = (key > qr) ? -1e30f : v;
      }

    // online softmax update per row (rows live in 16-lane halves)
#pragma unroll
    for (int r = 0; r < 8; ++r) {
      float t = fmaxf(fmaxf(s[0][r], s[1][r]), fmaxf(s[2][r], s[3][r]));
#pragma unroll
      for (int m = 8; m >= 1; m >>= 1) t = fmaxf(t, __shfl_xor(t, m, 32));
      const float mnew = fmaxf(mrun[r], t);
      const float corr = __expf(mrun[r] - mnew);
      float rs = 0.f;
#pragma unroll
      for (int kt2 = 0; kt2 < 4; ++kt2) {
        const float p = __expf(s[kt2][r] - mnew);
        s[kt2][r] = p;
        rs += p;
      }
#pragma unroll
      for (int m = 8; m >= 1; m >>= 1) rs += __shfl_xor(rs, m, 32);
      lrun[r] = lrun[r] * corr + rs;
      mrun[r] = mnew;
#pragma unroll
      for (int nt = 0; nt < 4; ++nt) o[nt][r] *= corr;
    }

    // P (C layout) -> LDS -> A-fragment layout
#pragma unroll
    for (int kt2 = 0; kt2 < 4; ++kt2)
#pragma unroll
      for (int r = 0; r < 8; ++r)
        sP[wave][mh + r][kt2 * 16 + nl] = (_Float16)s[kt2][r];
    __syncthreads();

    // O += P @ V  (keys 0..63 in two K=32 steps)
#pragma unroll
    for (int kh = 0; kh < 2; ++kh) {
      const v16h pa = load_a_frag(&sP[wave][0][kh * 32], 72, lane);
#pragma unroll
      for (int nt = 0; nt < 4; ++nt) {
        const v16h bv = load_b_frag(&sVt[nt * 16][kh * 32], 72, lane);
        o[nt] = wmma_f16(pa, bv, o[nt]);
      }
    }
  }

  // normalize and write back into (B,T,E) with head offset
#pragma unroll
  for (int nt = 0; nt < 4; ++nt)
#pragma unroll
    for (int r = 0; r < 8; ++r) {
      const int qr = qrow0 + mh + r;
      O[(baseBT + qr) * TE + col0 + nt * 16 + nl] = o[nt][r] / lrun[r];
    }
}

// ---------------------------------------------------------------------------
// Row-wise LayerNorm over E=1024, one workgroup per row, float4 per thread.
// ---------------------------------------------------------------------------
__global__ void __launch_bounds__(256)
ln_kernel(const float* __restrict__ in, const float* __restrict__ g,
          const float* __restrict__ b, float* __restrict__ out) {
  __shared__ float red[256];
  const int row = blockIdx.x;
  const int tid = threadIdx.x;
  const float4 xv = reinterpret_cast<const float4*>(in + (size_t)row * TE)[tid];

  red[tid] = xv.x + xv.y + xv.z + xv.w;
  __syncthreads();
  for (int st = 128; st > 0; st >>= 1) {
    if (tid < st) red[tid] += red[tid + st];
    __syncthreads();
  }
  const float mu = red[0] * (1.0f / TE);
  __syncthreads();

  const float dx = xv.x - mu, dy = xv.y - mu, dz = xv.z - mu, dw = xv.w - mu;
  red[tid] = dx * dx + dy * dy + dz * dz + dw * dw;
  __syncthreads();
  for (int st = 128; st > 0; st >>= 1) {
    if (tid < st) red[tid] += red[tid + st];
    __syncthreads();
  }
  const float rstd = rsqrtf(red[0] * (1.0f / TE) + 1e-5f);

  const float4 gv = reinterpret_cast<const float4*>(g)[tid];
  const float4 bv = reinterpret_cast<const float4*>(b)[tid];
  float4 ov;
  ov.x = dx * rstd * gv.x + bv.x;
  ov.y = dy * rstd * gv.y + bv.y;
  ov.z = dz * rstd * gv.z + bv.z;
  ov.w = dw * rstd * gv.w + bv.w;
  reinterpret_cast<float4*>(out + (size_t)row * TE)[tid] = ov;
}

// ---------------------------------------------------------------------------
// Embedding: x[b,t,:] = tok_emb[idx[b,t],:] + pos_emb[t,:]  (float4-wise)
// ---------------------------------------------------------------------------
__global__ void __launch_bounds__(256)
embed_kernel(const int* __restrict__ idx, const float* __restrict__ tok,
             const float* __restrict__ pos, float* __restrict__ x) {
  const size_t i = (size_t)blockIdx.x * 256 + threadIdx.x;  // float4 index
  if (i >= (size_t)TM * (TE / 4)) return;
  const int row = (int)(i / (TE / 4));
  const int e4  = (int)(i % (TE / 4));
  const int t   = row % TT;
  const float4 tv =
      reinterpret_cast<const float4*>(tok)[(size_t)idx[row] * (TE / 4) + e4];
  const float4 pv =
      reinterpret_cast<const float4*>(pos)[(size_t)t * (TE / 4) + e4];
  float4 ov;
  ov.x = tv.x + pv.x; ov.y = tv.y + pv.y;
  ov.z = tv.z + pv.z; ov.w = tv.w + pv.w;
  reinterpret_cast<float4*>(x)[i] = ov;
}

// ---------------------------------------------------------------------------
// Host side
// ---------------------------------------------------------------------------
extern "C" void kernel_launch(void* const* d_in, const int* in_sizes, int n_in,
                              void* d_out, int out_size, void* d_ws, size_t ws_size,
                              hipStream_t stream) {
  (void)in_sizes; (void)n_in; (void)out_size; (void)ws_size;

  const int*   idx     = (const int*)d_in[0];
  const float* tok_emb = (const float*)d_in[1];
  const float* pos_emb = (const float*)d_in[2];
  const float* Wq = (const float*)d_in[3];
  const float* bq = (const float*)d_in[4];
  const float* Wk = (const float*)d_in[5];
  const float* bk = (const float*)d_in[6];
  const float* Wv = (const float*)d_in[7];
  const float* bv = (const float*)d_in[8];
  const float* Wo = (const float*)d_in[9];
  const float* bo = (const float*)d_in[10];
  const float* W1 = (const float*)d_in[11];
  const float* b1 = (const float*)d_in[12];
  const float* W2 = (const float*)d_in[13];
  const float* b2 = (const float*)d_in[14];
  const float* ln1_g = (const float*)d_in[15];
  const float* ln1_b = (const float*)d_in[16];
  const float* ln2_g = (const float*)d_in[17];
  const float* ln2_b = (const float*)d_in[18];
  const float* lnf_g = (const float*)d_in[19];
  const float* lnf_b = (const float*)d_in[20];
  const float* Whead = (const float*)d_in[21];
  const float* bhead = (const float*)d_in[22];

  // workspace layout (f32): [x][q][k][v][attn][ffo]; ffh aliases q..attn (4*ME).
  const size_t ME = (size_t)TM * TE;
  float* ws   = (float*)d_ws;
  float* x    = ws;
  float* q    = ws + 1 * ME;
  float* k    = ws + 2 * ME;
  float* v    = ws + 3 * ME;
  float* attn = ws + 4 * ME;
  float* ffh  = q;               // TM x TFF == 4*ME elements
  float* ffo  = ws + 5 * ME;

  const dim3 blk(256);
  const dim3 gEE(TE / 128, TM / 256);     // (8, 16)
  const dim3 gFF(TFF / 128, TM / 256);    // (32, 16)
  const dim3 gHead(TV / 128, TM / 256);   // (250, 16)
  const dim3 gAttn(TB * THN, TT / 64);    // (64, 16)

  embed_kernel<<<(unsigned)((TM * (TE / 4) + 255) / 256), blk, 0, stream>>>(
      idx, tok_emb, pos_emb, x);

  for (int l = 0; l < TL; ++l) {
    const float* Wq_l = Wq + (size_t)l * TE * TE;
    const float* Wk_l = Wk + (size_t)l * TE * TE;
    const float* Wv_l = Wv + (size_t)l * TE * TE;
    const float* Wo_l = Wo + (size_t)l * TE * TE;
    const float* W1_l = W1 + (size_t)l * TE * TFF;
    const float* W2_l = W2 + (size_t)l * TFF * TE;
    const float* bq_l = bq + (size_t)l * TE;
    const float* bk_l = bk + (size_t)l * TE;
    const float* bv_l = bv + (size_t)l * TE;
    const float* bo_l = bo + (size_t)l * TE;
    const float* b1_l = b1 + (size_t)l * TFF;
    const float* b2_l = b2 + (size_t)l * TE;

    // QKV projections
    gemm_wmma<<<gEE, blk, 0, stream>>>(x, Wq_l, bq_l, nullptr, q, TM, TE, TE, 1);
    gemm_wmma<<<gEE, blk, 0, stream>>>(x, Wk_l, bk_l, nullptr, k, TM, TE, TE, 1);
    gemm_wmma<<<gEE, blk, 0, stream>>>(x, Wv_l, bv_l, nullptr, v, TM, TE, TE, 1);

    // causal flash attention
    attn_kernel<<<gAttn, dim3(128), 0, stream>>>(q, k, v, attn);

    // output projection + bias + residual -> k (scratch), then LN1 -> x
    gemm_wmma<<<gEE, blk, 0, stream>>>(attn, Wo_l, bo_l, x, k, TM, TE, TE, 1 | 4);
    ln_kernel<<<TM, blk, 0, stream>>>(k, ln1_g + (size_t)l * TE,
                                      ln1_b + (size_t)l * TE, x);

    // FFN: relu(x@W1+b1) @ W2 + b2 + residual, then LN2 -> x
    gemm_wmma<<<gFF, blk, 0, stream>>>(x, W1_l, b1_l, nullptr, ffh, TM, TFF, TE, 1 | 2);
    gemm_wmma<<<gEE, blk, 0, stream>>>(ffh, W2_l, b2_l, x, ffo, TM, TE, TFF, 1 | 4);
    ln_kernel<<<TM, blk, 0, stream>>>(ffo, ln2_g + (size_t)l * TE,
                                      ln2_b + (size_t)l * TE, x);
  }

  // final LN -> q (scratch), then logits head -> d_out
  ln_kernel<<<TM, blk, 0, stream>>>(x, lnf_g, lnf_b, q);
  gemm_wmma<<<gHead, blk, 0, stream>>>(q, Whead, bhead, nullptr, (float*)d_out,
                                       TM, TV, TE, 1);
}
